// InfoNCELoss_39943195853108
// MI455X (gfx1250) — matrix operand order repaired and checked by reading, not compiled, not run on hardware
//
#include <hip/hip_runtime.h>
#include <hip/hip_bf16.h>

typedef __attribute__((ext_vector_type(16))) _Float16 v16h;
typedef __attribute__((ext_vector_type(8)))  float    v8f;
typedef __attribute__((ext_vector_type(4)))  _Float16 v4h;
typedef __attribute__((ext_vector_type(4)))  unsigned int v4u;
typedef __attribute__((ext_vector_type(8)))  int      v8i_;
typedef __attribute__((ext_vector_type(4)))  int      v4i_;

#define NROWS 16384
#define DIM   128
#define INV_T (1.0f / 0.07f)

union U16h { uint4 q[2]; v16h h; };

// ---------------------------------------------------------------------------
// Phase 1: L2-normalize rows of A and P to f16, compute fp32 diagonal logits.
// One wave per row: 32 lanes x 4 floats = 128.
// ---------------------------------------------------------------------------
__global__ __launch_bounds__(256)
void nce_normalize_kernel(const float* __restrict__ A, const float* __restrict__ P,
                          _Float16* __restrict__ Ah, _Float16* __restrict__ Ph,
                          float* __restrict__ diag) {
    const int wave = threadIdx.x >> 5;
    const int lane = threadIdx.x & 31;
    const int row  = blockIdx.x * 8 + wave;

    const float4 a = ((const float4*)(A + (size_t)row * DIM))[lane];
    const float4 p = ((const float4*)(P + (size_t)row * DIM))[lane];

    float na2 = a.x*a.x + a.y*a.y + a.z*a.z + a.w*a.w;
    float np2 = p.x*p.x + p.y*p.y + p.z*p.z + p.w*p.w;
    float dp  = a.x*p.x + a.y*p.y + a.z*p.z + a.w*p.w;
    #pragma unroll
    for (int off = 16; off > 0; off >>= 1) {
        na2 += __shfl_xor(na2, off, 32);
        np2 += __shfl_xor(np2, off, 32);
        dp  += __shfl_xor(dp,  off, 32);
    }
    const float na = fmaxf(sqrtf(na2), 1e-12f);
    const float np = fmaxf(sqrtf(np2), 1e-12f);
    const float ra = 1.0f / na, rp = 1.0f / np;

    v4h ah, ph;
    ah[0] = (_Float16)(a.x * ra); ah[1] = (_Float16)(a.y * ra);
    ah[2] = (_Float16)(a.z * ra); ah[3] = (_Float16)(a.w * ra);
    ph[0] = (_Float16)(p.x * rp); ph[1] = (_Float16)(p.y * rp);
    ph[2] = (_Float16)(p.z * rp); ph[3] = (_Float16)(p.w * rp);
    ((v4h*)(Ah + (size_t)row * DIM))[lane] = ah;
    ((v4h*)(Ph + (size_t)row * DIM))[lane] = ph;

    if (lane == 0) diag[row] = dp * ra * rp * INV_T;  // exact fp32 diagonal logit
}

// ---------------------------------------------------------------------------
// Phase 2: fused GEMM + sum(exp(logit - 1/T)) per row via WMMA f32_16x16x32_f16.
// B tiles staged with the Tensor Data Mover (double-buffered, overlaps compute).
// Block = 256 thr = 8 waves = 4 row-strips x 2 column-parities; 64 rows/block.
// ---------------------------------------------------------------------------
#define LDS_STRIDE 136                 // 128 halfs + 8 pad (TDM pad: 16B per 256B row)
#define TILE_HALFS (16 * LDS_STRIDE)   // one 16x128 subtile
#define BUF_HALFS  (2 * TILE_HALFS)    // one TDM fetch = 32 rows = 2 subtiles

// Issue a TDM load of a 32x128 f16 tile (rows of Ph) into LDS at ldsByteAddr.
// D# per CDNA5 ISA ch.8: 2-D tensor, data_size=2B, hw row padding 4 DW / 64 DW.
__device__ __forceinline__ void tdm_load_tile(unsigned int ldsByteAddr,
                                              unsigned long long gaddr) {
    v4u g0;
    g0[0] = 1u;                                        // count=1, user mode
    g0[1] = ldsByteAddr;                               // lds_addr [63:32]
    g0[2] = (unsigned int)(gaddr & 0xFFFFFFFFu);       // global_addr lo
    g0[3] = (unsigned int)(gaddr >> 32) | 0x80000000u; // global_addr hi | type=2

    v8i_ g1;
    g1[0] = (int)((1u << 16)    // data_size = 1 -> 2 bytes
                | (1u << 20)    // pad_enable
                | (5u << 22)    // pad_interval = 5 -> every 64 DWORDs (256B row)
                | (3u << 25));  // pad_amount   = 3 -> 4 DWORDs (16B)
    g1[1] = (int)(128u << 16);   // tensor_dim0 = 128 (bits 79:48, low half here)
    g1[2] = (int)(16384u << 16); // tensor_dim0 hi=0 | tensor_dim1 lo16 = 16384
    g1[3] = (int)(128u << 16);   // tensor_dim1 hi=0 | tile_dim0 = 128
    g1[4] = 32;                  // tile_dim1 = 32 rows, tile_dim2 = 0
    g1[5] = 128;                 // tensor_dim0_stride = 128 (lo32)
    g1[6] = 0;                   // stride hi, dim1_stride lo
    g1[7] = 0;

    v4i_ gz4 = {0, 0, 0, 0};               // groups 2/3: unused (2-D tensor)
    v8i_ gz8 = {0, 0, 0, 0, 0, 0, 0, 0};   // extra group slot (clang-23 6-arg form)
    __builtin_amdgcn_tensor_load_to_lds(g0, g1, gz4, gz4, gz8, 0);
}

__global__ __launch_bounds__(256)
void nce_lse_kernel(const _Float16* __restrict__ Ah, const _Float16* __restrict__ Ph,
                    float* __restrict__ rowsum) {
    __shared__ __align__(16) _Float16 smem[2 * BUF_HALFS];  // double buffer, 17408 B
    __shared__ float sRow[64];

    const int wave = threadIdx.x >> 5;
    const int lane = threadIdx.x & 31;
    const int lg   = lane >> 4;     // lane half 0/1
    const int lr   = lane & 15;     // lane within half
    const int strip  = wave & 3;    // which 16-row strip
    const int parity = wave >> 2;   // which column subtile of the 32-row fetch
    const int rowBase = blockIdx.x * 64 + strip * 16;
    const bool isWave0 = (wave == 0);

    if (threadIdx.x < 64) sRow[threadIdx.x] = 0.0f;

    const unsigned int ldsBase = (unsigned int)(size_t)(&smem[0]);
    const unsigned long long pBase = (unsigned long long)(size_t)Ph;

    // Preload A fragments (4 K-chunks of 32), ISA 16-bit A layout:
    //   lane<16 holds K {0..7} and {16..23}; lane>=16 holds K {8..15} and {24..31}
    U16h aF[4];
    const _Float16* arow = Ah + (size_t)(rowBase + lr) * DIM;
    #pragma unroll
    for (int c = 0; c < 4; ++c) {
        aF[c].q[0] = *(const uint4*)(arow + 32 * c + 8 * lg);
        aF[c].q[1] = *(const uint4*)(arow + 32 * c + 16 + 8 * lg);
    }

    float sums[8] = {0.f, 0.f, 0.f, 0.f, 0.f, 0.f, 0.f, 0.f};

    constexpr int NIT = (NROWS / 32);        // 512 iterations of 32 columns
    // Prologue: kick off tile 0 into buffer 0.
    if (isWave0) tdm_load_tile(ldsBase, pBase);

    int cur = 0;
    for (int it = 0; it < NIT; ++it) {
        if (isWave0) {
            __builtin_amdgcn_s_wait_tensorcnt(0);     // buf[cur] arrived
        }
        __syncthreads();  // all: buf[cur] ready; buf[cur^1] consumption (it-1) done
        if (isWave0 && (it + 1) < NIT) {
            tdm_load_tile(ldsBase + (unsigned int)((cur ^ 1) * BUF_HALFS * 2),
                          pBase + (unsigned long long)(it + 1) * 32u * DIM * 2u);
        }

        // Consume buf[cur]: this wave's 16x16 output tile, K=128 in 4 WMMAs.
        const _Float16* tbase = smem + cur * BUF_HALFS + parity * TILE_HALFS;
        U16h bF[4];
        #pragma unroll
        for (int c = 0; c < 4; ++c) {
            // B layout: lane<16 -> K {0..7},{16..23}; lane>=16 -> K {8..15},{24..31}
            const _Float16* bsrc = tbase + lr * LDS_STRIDE + 32 * c + 8 * lg;
            bF[c].q[0] = *(const uint4*)(bsrc);
            bF[c].q[1] = *(const uint4*)(bsrc + 16);
        }
        v8f acc = {};
        #pragma unroll
        for (int c = 0; c < 4; ++c)
            acc = __builtin_amdgcn_wmma_f32_16x16x32_f16(
                false, aF[c].h, false, bF[c].h, (short)0, acc, false, false);

        // logits <= 1/T (unit vectors): exp((dot-1)/T) can never overflow.
        #pragma unroll
        for (int r = 0; r < 8; ++r)
            sums[r] += __expf((acc[r] - 1.0f) * INV_T);

        __syncthreads();  // consumption of buf[cur] done before it's overwritten
        cur ^= 1;
    }

    // Row-sum: D element (VGPR r, lane l) is row r+8*(l>=16), col l%16.
    #pragma unroll
    for (int r = 0; r < 8; ++r) {
        float s = sums[r];
        s += __shfl_xor(s, 1, 32);
        s += __shfl_xor(s, 2, 32);
        s += __shfl_xor(s, 4, 32);
        s += __shfl_xor(s, 8, 32);
        sums[r] = s;
    }
    if (lr == 0) {  // lane 0 -> local rows 0..7, lane 16 -> rows 8..15
        #pragma unroll
        for (int r = 0; r < 8; ++r)
            atomicAdd(&sRow[strip * 16 + lg * 8 + r], sums[r]);  // combine 2 parities
    }
    __syncthreads();
    if (threadIdx.x < 64)
        rowsum[blockIdx.x * 64 + threadIdx.x] = sRow[threadIdx.x];
}

// ---------------------------------------------------------------------------
// Phase 3: loss = mean( log(S_i) + 1/T - diag_i )
// ---------------------------------------------------------------------------
__global__ __launch_bounds__(256)
void nce_finalize_kernel(const float* __restrict__ rowsum,
                         const float* __restrict__ diag,
                         float* __restrict__ out) {
    __shared__ float red[256];
    float acc = 0.0f;
    for (int i = threadIdx.x; i < NROWS; i += 256)
        acc += logf(rowsum[i]) + INV_T - diag[i];
    red[threadIdx.x] = acc;
    __syncthreads();
    #pragma unroll
    for (int s = 128; s > 0; s >>= 1) {
        if (threadIdx.x < s) red[threadIdx.x] += red[threadIdx.x + s];
        __syncthreads();
    }
    if (threadIdx.x == 0) out[0] = red[0] / (float)NROWS;
}

extern "C" void kernel_launch(void* const* d_in, const int* in_sizes, int n_in,
                              void* d_out, int out_size, void* d_ws, size_t ws_size,
                              hipStream_t stream) {
    const float* A = (const float*)d_in[0];
    const float* P = (const float*)d_in[1];

    char* ws = (char*)d_ws;
    const size_t halfMat = (size_t)NROWS * DIM * sizeof(_Float16);  // 4 MB
    _Float16* Ah     = (_Float16*)(ws);
    _Float16* Ph     = (_Float16*)(ws + halfMat);
    float*    rowsum = (float*)(ws + 2 * halfMat);
    float*    diag   = (float*)(ws + 2 * halfMat + (size_t)NROWS * sizeof(float));

    nce_normalize_kernel<<<NROWS / 8, 256, 0, stream>>>(A, P, Ah, Ph, diag);
    nce_lse_kernel<<<NROWS / 64, 256, 0, stream>>>(Ah, Ph, rowsum);
    nce_finalize_kernel<<<1, 256, 0, stream>>>(rowsum, diag, (float*)d_out);
}
